// CausalMultiHeadSelfAttention_88837103551553
// MI455X (gfx1250) — matrix operand order repaired and checked by reading, not compiled, not run on hardware
//
#include <hip/hip_runtime.h>

typedef __bf16 bf16_t;
typedef __attribute__((ext_vector_type(16))) __bf16 v16bf;
typedef __attribute__((ext_vector_type(8)))  __bf16 v8bf;
typedef __attribute__((ext_vector_type(8)))  float  v8f;

#define DEV __device__ __forceinline__

constexpr int BATCH  = 4;
constexpr int SEQ    = 2048;
constexpr int DMODEL = 1024;
constexpr int HEADS  = 16;
constexpr int DHEAD  = 64;
constexpr int MROWS  = BATCH * SEQ;   // 8192

// ln(10000)/64
constexpr float ROPE_LN_SCALE = 0.14391156f;

// ---------------------------------------------------------------------------
// WMMA helpers (CDNA5 wave32, v_wmma_f32_16x16x32_bf16)
// ---------------------------------------------------------------------------
DEV v8f wmma_bf16(v16bf a, v16bf b, v8f c) {
  // (neg_a, A, neg_b, B, c_mod, C, reuse_a, reuse_b)
  return __builtin_amdgcn_wmma_f32_16x16x32_bf16(false, a, false, b,
                                                 (short)0, c, false, false);
}

// A-matrix 16x32 bf16 fragment. ISA layout:
//   lanes 0-15 : halves[0:8]=K0..7,   halves[8:16]=K16..23  (row M = lane)
//   lanes 16-31: halves[0:8]=K8..15,  halves[8:16]=K24..31  (row M = lane-16)
DEV v16bf make_afrag(const bf16_t* __restrict__ row, int k0, int lane) {
  const int off = (lane & 16) ? 8 : 0;
  v8bf lo = *(const v8bf*)(row + k0 + off);
  v8bf hi = *(const v8bf*)(row + k0 + 16 + off);
  return __builtin_shufflevector(lo, hi, 0, 1, 2, 3, 4, 5, 6, 7,
                                         8, 9, 10, 11, 12, 13, 14, 15);
}

// B-matrix 32x16 bf16 fragment: lane (n = lane&15) holds 16 contraction
// values of its column; lanes 0-15: K=k0..k0+7, k0+16..k0+23,
// lanes 16-31: K=k0+8..k0+15, k0+24..k0+31.  `col` points at the column's K-major data.
DEV v16bf make_bfrag(const bf16_t* __restrict__ col, int k0, int lane) {
  const int off = (lane & 16) ? 8 : 0;
  v8bf lo = *(const v8bf*)(col + k0 + off);
  v8bf hi = *(const v8bf*)(col + k0 + 16 + off);
  return __builtin_shufflevector(lo, hi, 0, 1, 2, 3, 4, 5, 6, 7,
                                         8, 9, 10, 11, 12, 13, 14, 15);
}

// ---------------------------------------------------------------------------
// fp32 -> bf16 conversion
// ---------------------------------------------------------------------------
__global__ void f32_to_bf16_kernel(const float* __restrict__ in,
                                   bf16_t* __restrict__ out, int n) {
  int i = blockIdx.x * blockDim.x + threadIdx.x;
  if (i < n) out[i] = (bf16_t)in[i];
}

// ---------------------------------------------------------------------------
// Fused QKV projection + RoPE.
//   q = rope(x @ Wq^T), k = rope(x @ Wk^T), v = x @ Wv^T
// One wave computes a 16(M) x 32(N) tile for each of the three outputs,
// reusing every A-fragment across 6 WMMAs.  RoPE is applied in f32 on the
// accumulators: pairs (2t,2t+1) live in adjacent lanes -> shfl_xor(.,1).
// Outputs: q/k as [B,H,S,64] bf16, v transposed as [B,H,64,S] bf16.
// ---------------------------------------------------------------------------
__global__ void __launch_bounds__(128)
qkv_rope_kernel(const bf16_t* __restrict__ A,
                const bf16_t* __restrict__ Wq, const bf16_t* __restrict__ Wk,
                const bf16_t* __restrict__ Wv, bf16_t* __restrict__ qout,
                bf16_t* __restrict__ kout, bf16_t* __restrict__ vtout,
                const int* __restrict__ tok) {
  const int lane = threadIdx.x & 31;
  const int wave = blockIdx.x * 4 + (threadIdx.x >> 5);
  const int tilesN = DMODEL / 32;               // 32
  const int m0 = (wave / tilesN) * 16;
  const int n0 = (wave % tilesN) * 32;
  const int n_lane = lane & 15;

  const bf16_t* arow = A + (size_t)(m0 + n_lane) * DMODEL;
  const bf16_t* wq0 = Wq + (size_t)(n0 + n_lane) * DMODEL;
  const bf16_t* wq1 = Wq + (size_t)(n0 + 16 + n_lane) * DMODEL;
  const bf16_t* wk0 = Wk + (size_t)(n0 + n_lane) * DMODEL;
  const bf16_t* wk1 = Wk + (size_t)(n0 + 16 + n_lane) * DMODEL;
  const bf16_t* wv0 = Wv + (size_t)(n0 + n_lane) * DMODEL;
  const bf16_t* wv1 = Wv + (size_t)(n0 + 16 + n_lane) * DMODEL;

  v8f cq[2] = {v8f{}, v8f{}};
  v8f ck[2] = {v8f{}, v8f{}};
  v8f cv[2] = {v8f{}, v8f{}};
  for (int k0 = 0; k0 < DMODEL; k0 += 32) {
    // speculative prefetch of the next K-slab (global_prefetch_b8)
    __builtin_prefetch((const void*)(arow + k0 + 256), 0, 1);
    __builtin_prefetch((const void*)(wq0 + k0 + 256), 0, 1);
    v16bf a = make_afrag(arow, k0, lane);
    cq[0] = wmma_bf16(a, make_bfrag(wq0, k0, lane), cq[0]);
    cq[1] = wmma_bf16(a, make_bfrag(wq1, k0, lane), cq[1]);
    ck[0] = wmma_bf16(a, make_bfrag(wk0, k0, lane), ck[0]);
    ck[1] = wmma_bf16(a, make_bfrag(wk1, k0, lane), ck[1]);
    cv[0] = wmma_bf16(a, make_bfrag(wv0, k0, lane), cv[0]);
    cv[1] = wmma_bf16(a, make_bfrag(wv1, k0, lane), cv[1]);
  }

  // Epilogue. C/D layout: lane holds column n = lane&15; VGPR r holds row
  // r (+8 for lanes 16-31).
  const int mbase = (lane & 16) ? 8 : 0;
#pragma unroll
  for (int r = 0; r < 8; ++r) {
    const int m = m0 + mbase + r;
    const int bb = m >> 11;                 // m / SEQ
    const int s = m & (SEQ - 1);
    const float pos = (float)tok[s];
#pragma unroll
    for (int g = 0; g < 2; ++g) {
      const int n = n0 + g * 16 + n_lane;
      const int h = n >> 6;
      const int d = n & 63;
      const int t = d >> 1;
      // RoPE angle for this pair
      const float invf = __expf(-(float)(2 * t) * ROPE_LN_SCALE);
      const float ang = pos * invf;
      const float cs = cosf(ang);
      const float sn = sinf(ang);

      const float vq = cq[g][r];
      const float vk = ck[g][r];
      const float vv = cv[g][r];
      const float pq = __shfl_xor(vq, 1, 32);   // partner within RoPE pair
      const float pk = __shfl_xor(vk, 1, 32);
      const bool odd = (d & 1) != 0;
      const float rq = odd ? (pq * sn + vq * cs) : (vq * cs - pq * sn);
      const float rk = odd ? (pk * sn + vk * cs) : (vk * cs - pk * sn);

      const size_t qkidx = ((size_t)(bb * HEADS + h) * SEQ + s) * DHEAD + d;
      qout[qkidx] = (bf16_t)rq;
      kout[qkidx] = (bf16_t)rk;
      vtout[((size_t)(bb * HEADS + h) * DHEAD + d) * SEQ + s] = (bf16_t)vv;
    }
  }
}

// ---------------------------------------------------------------------------
// Output projection: out = Ob[8192,1024] @ Wo[1024,1024]^T -> fp32.
// One wave computes a 16(M) x 64(N) tile.
// ---------------------------------------------------------------------------
__global__ void __launch_bounds__(128)
out_proj_kernel(const bf16_t* __restrict__ A, const bf16_t* __restrict__ W,
                float* __restrict__ out) {
  const int lane = threadIdx.x & 31;
  const int wave = blockIdx.x * 4 + (threadIdx.x >> 5);
  const int tilesN = DMODEL / 64;               // 16
  const int m0 = (wave / tilesN) * 16;
  const int n0 = (wave % tilesN) * 64;
  const int n_lane = lane & 15;

  const bf16_t* arow = A + (size_t)(m0 + n_lane) * DMODEL;
  const bf16_t* w0 = W + (size_t)(n0 + 0 * 16 + n_lane) * DMODEL;
  const bf16_t* w1 = W + (size_t)(n0 + 1 * 16 + n_lane) * DMODEL;
  const bf16_t* w2 = W + (size_t)(n0 + 2 * 16 + n_lane) * DMODEL;
  const bf16_t* w3 = W + (size_t)(n0 + 3 * 16 + n_lane) * DMODEL;

  v8f c0{}, c1{}, c2{}, c3{};
  for (int k0 = 0; k0 < DMODEL; k0 += 32) {
    __builtin_prefetch((const void*)(arow + k0 + 256), 0, 1);
    __builtin_prefetch((const void*)(w0 + k0 + 256), 0, 1);
    v16bf a = make_afrag(arow, k0, lane);
    c0 = wmma_bf16(a, make_bfrag(w0, k0, lane), c0);
    c1 = wmma_bf16(a, make_bfrag(w1, k0, lane), c1);
    c2 = wmma_bf16(a, make_bfrag(w2, k0, lane), c2);
    c3 = wmma_bf16(a, make_bfrag(w3, k0, lane), c3);
  }

  const int mbase = (lane & 16) ? 8 : 0;
#pragma unroll
  for (int r = 0; r < 8; ++r) {
    const size_t m = m0 + mbase + r;
    out[m * DMODEL + n0 + 0 * 16 + n_lane] = c0[r];
    out[m * DMODEL + n0 + 1 * 16 + n_lane] = c1[r];
    out[m * DMODEL + n0 + 2 * 16 + n_lane] = c2[r];
    out[m * DMODEL + n0 + 3 * 16 + n_lane] = c3[r];
  }
}

// ---------------------------------------------------------------------------
// Flash attention: one wave owns 16 queries of one (b,h); loops over 32-key
// blocks with online softmax. qb/kb: [B,H,S,64] bf16 (RoPE applied),
// vt: [B,H,64,S] bf16, Ob: [B,S,1024] bf16.
// ---------------------------------------------------------------------------
__global__ void __launch_bounds__(128)
flash_attn_kernel(const bf16_t* __restrict__ qb, const bf16_t* __restrict__ kb,
                  const bf16_t* __restrict__ vt, bf16_t* __restrict__ Ob) {
  __shared__ alignas(16) bf16_t Pbuf[4][16][40];  // per-wave P tile, padded

  const int lane = threadIdx.x & 31;
  const int wid = threadIdx.x >> 5;
  const int wave = blockIdx.x * 4 + wid;
  const int qt = wave & (SEQ / 16 - 1);       // query tile in [0,128)
  const int bh = wave >> 7;
  const int q0 = qt * 16;

  const bf16_t* qbase = qb + (size_t)bh * SEQ * DHEAD;
  const bf16_t* kbase = kb + (size_t)bh * SEQ * DHEAD;
  const bf16_t* vbase = vt + (size_t)bh * DHEAD * SEQ;

  const int n_lane = lane & 15;
  const int half = (lane & 16) ? 1 : 0;

  // Q fragments (16 queries x 64 dims = two K-chunks of 32)
  const bf16_t* qrow = qbase + (size_t)(q0 + n_lane) * DHEAD;
  const v16bf qa0 = make_afrag(qrow, 0, lane);
  const v16bf qa1 = make_afrag(qrow, 32, lane);

  float mrow[8], lrow[8];
  v8f o0{}, o1{}, o2{}, o3{};
#pragma unroll
  for (int r = 0; r < 8; ++r) { mrow[r] = -3.0e38f; lrow[r] = 0.0f; }

  const int nb = (q0 + 16 + 31) >> 5;   // number of 32-key blocks (causal)
  for (int j = 0; j < nb; ++j) {
    const int kb0 = j * 32;

    // --- scores: 16 q x 32 keys, contraction over 64 head dims -----------
    const bf16_t* krow0 = kbase + (size_t)(kb0 + n_lane) * DHEAD;   // keys g=0
    const bf16_t* krow1 = krow0 + (size_t)16 * DHEAD;               // keys g=1
    v8f z{};
    v8f s0 = wmma_bf16(qa0, make_bfrag(krow0, 0, lane), z);
    s0 = wmma_bf16(qa1, make_bfrag(krow0, 32, lane), s0);
    v8f s1 = wmma_bf16(qa0, make_bfrag(krow1, 0, lane), z);
    s1 = wmma_bf16(qa1, make_bfrag(krow1, 32, lane), s1);

    // --- online softmax (rows reduced across 16 lanes of each half) ------
#pragma unroll
    for (int r = 0; r < 8; ++r) {
      const int m_local = r + 8 * half;
      const int sq = q0 + m_local;
      float e0 = s0[r] * 0.125f;
      if (kb0 + n_lane > sq) e0 = -1.0e9f;
      float e1 = s1[r] * 0.125f;
      if (kb0 + 16 + n_lane > sq) e1 = -1.0e9f;

      float mx = fmaxf(e0, e1);
#pragma unroll
      for (int d = 1; d < 16; d <<= 1) mx = fmaxf(mx, __shfl_xor(mx, d, 32));
      const float mnew = fmaxf(mrow[r], mx);
      const float alpha = __expf(mrow[r] - mnew);
      const float p0 = __expf(e0 - mnew);
      const float p1 = __expf(e1 - mnew);
      float ps = p0 + p1;
#pragma unroll
      for (int d = 1; d < 16; d <<= 1) ps += __shfl_xor(ps, d, 32);
      lrow[r] = lrow[r] * alpha + ps;
      mrow[r] = mnew;
      o0[r] *= alpha; o1[r] *= alpha; o2[r] *= alpha; o3[r] *= alpha;

      Pbuf[wid][m_local][n_lane]      = (bf16_t)p0;
      Pbuf[wid][m_local][n_lane + 16] = (bf16_t)p1;
    }

    // DS ops are in-order per wave: drain stores before the vector reloads.
    asm volatile("s_wait_dscnt 0" ::: "memory");

    // --- reload P in A-fragment layout (16x32) ---------------------------
    const bf16_t* prow = &Pbuf[wid][n_lane][0];
    const int aoff = half ? 8 : 0;
    v8bf plo = *(const v8bf*)(prow + aoff);
    v8bf phi = *(const v8bf*)(prow + 16 + aoff);
    const v16bf pa = __builtin_shufflevector(plo, phi, 0, 1, 2, 3, 4, 5, 6, 7,
                                             8, 9, 10, 11, 12, 13, 14, 15);

    // --- out += P (16x32) @ V (32 keys x 64 dims), V is [64][S] ----------
    const bf16_t* v0 = vbase + (size_t)(0 * 16 + n_lane) * SEQ;
    const bf16_t* v1 = vbase + (size_t)(1 * 16 + n_lane) * SEQ;
    const bf16_t* v2 = vbase + (size_t)(2 * 16 + n_lane) * SEQ;
    const bf16_t* v3 = vbase + (size_t)(3 * 16 + n_lane) * SEQ;
    o0 = wmma_bf16(pa, make_bfrag(v0, kb0, lane), o0);
    o1 = wmma_bf16(pa, make_bfrag(v1, kb0, lane), o1);
    o2 = wmma_bf16(pa, make_bfrag(v2, kb0, lane), o2);
    o3 = wmma_bf16(pa, make_bfrag(v3, kb0, lane), o3);
  }

  // --- epilogue: normalize and scatter into [B,S,H*64] ---------------------
  const int b = bh >> 4;
  const int h = bh & 15;
#pragma unroll
  for (int r = 0; r < 8; ++r) {
    const int s = q0 + r + 8 * half;
    const float inv = 1.0f / lrow[r];
    const size_t base = ((size_t)(b * SEQ + s)) * DMODEL + h * DHEAD + n_lane;
    Ob[base + 0]  = (bf16_t)(o0[r] * inv);
    Ob[base + 16] = (bf16_t)(o1[r] * inv);
    Ob[base + 32] = (bf16_t)(o2[r] * inv);
    Ob[base + 48] = (bf16_t)(o3[r] * inv);
  }
}

// ---------------------------------------------------------------------------
// Host-side orchestration
// ---------------------------------------------------------------------------
extern "C" void kernel_launch(void* const* d_in, const int* in_sizes, int n_in,
                              void* d_out, int out_size, void* d_ws,
                              size_t ws_size, hipStream_t stream) {
  const float* x  = (const float*)d_in[0];
  const int* tok  = (const int*)d_in[1];
  const float* Wq = (const float*)d_in[2];
  const float* Wk = (const float*)d_in[3];
  const float* Wv = (const float*)d_in[4];
  const float* Wo = (const float*)d_in[5];

  char* ws = (char*)d_ws;
  const size_t MB = 1024ull * 1024ull;
  bf16_t* Xb   = (bf16_t*)(ws);              // 16 MB : x in bf16; reused as Ob
  bf16_t* Wqb  = (bf16_t*)(ws + 16 * MB);    //  2 MB
  bf16_t* Wkb  = (bf16_t*)(ws + 18 * MB);    //  2 MB
  bf16_t* Wvb  = (bf16_t*)(ws + 20 * MB);    //  2 MB
  bf16_t* Wob  = (bf16_t*)(ws + 22 * MB);    //  2 MB
  bf16_t* qbuf = (bf16_t*)(ws + 24 * MB);    // 16 MB [B,H,S,64], rope applied
  bf16_t* kbuf = (bf16_t*)(ws + 40 * MB);    // 16 MB [B,H,S,64], rope applied
  bf16_t* vtb  = (bf16_t*)(ws + 56 * MB);    // 16 MB [B,H,64,S]
  bf16_t* Obuf = Xb;                         // reuse: x-bf16 dead after QKV

  // 1) convert inputs to bf16
  const int nx = BATCH * SEQ * DMODEL;       // 8388608
  const int nw = DMODEL * DMODEL;            // 1048576
  f32_to_bf16_kernel<<<(nx + 255) / 256, 256, 0, stream>>>(x, Xb, nx);
  f32_to_bf16_kernel<<<(nw + 255) / 256, 256, 0, stream>>>(Wq, Wqb, nw);
  f32_to_bf16_kernel<<<(nw + 255) / 256, 256, 0, stream>>>(Wk, Wkb, nw);
  f32_to_bf16_kernel<<<(nw + 255) / 256, 256, 0, stream>>>(Wv, Wvb, nw);
  f32_to_bf16_kernel<<<(nw + 255) / 256, 256, 0, stream>>>(Wo, Wob, nw);

  // 2) fused QKV projection + RoPE:
  //    (M/16)*(N/32) = 512*32 = 16384 waves -> 4096 blocks of 4 waves
  qkv_rope_kernel<<<16384 / 4, 128, 0, stream>>>(Xb, Wqb, Wkb, Wvb, qbuf, kbuf,
                                                 vtb, tok);

  // 3) causal flash attention: B*H*(S/16) = 8192 waves -> 2048 blocks
  flash_attn_kernel<<<8192 / 4, 128, 0, stream>>>(qbuf, kbuf, vtb, Obuf);

  // 4) output projection -> fp32 d_out:
  //    (M/16)*(N/64) = 512*16 = 8192 waves -> 2048 blocks
  out_proj_kernel<<<8192 / 4, 128, 0, stream>>>(Obuf, Wob, (float*)d_out);
}
